// MAB_16355235463369
// MI455X (gfx1250) — matrix operand order repaired and checked by reading, not compiled, not run on hardware
//
#include <hip/hip_runtime.h>
#include <hip/hip_bf16.h>
#include <stdint.h>

// ---------------- problem constants ----------------
constexpr int Bc  = 8;
constexpr int NXc = 1024;
constexpr int NYc = 1024;
constexpr int DIMc = 1024;
constexpr int Hc  = 8;
constexpr int DHc = 128;              // head dim
constexpr int Mrows = Bc * NXc;       // 8192 rows for the big GEMMs
#define NEG_INF (-__builtin_inff())

// ---------------- WMMA types / helpers ----------------
typedef __bf16    bf16x16 __attribute__((ext_vector_type(16)));
typedef float     v8f     __attribute__((ext_vector_type(8)));
typedef uint32_t  u32x4   __attribute__((ext_vector_type(4)));

__device__ __forceinline__ uint16_t f2bf(float x) {
    uint32_t u = __float_as_uint(x);
    uint32_t r = (u + 0x7FFFu + ((u >> 16) & 1u)) >> 16;   // round-to-nearest-even
    return (uint16_t)r;
}
__device__ __forceinline__ uint32_t pack_bf2(float lo, float hi) {
    return (uint32_t)f2bf(lo) | ((uint32_t)f2bf(hi) << 16);
}

__device__ __forceinline__ v8f wmma_bf16(const bf16x16& a, const bf16x16& b, v8f c) {
    return __builtin_amdgcn_wmma_f32_16x16x32_bf16(
        /*neg_a=*/false, a, /*neg_b=*/false, b,
        /*c_mod=*/(short)0, c, /*reuse_a=*/false, /*reuse_b=*/false);
}

// A-fragment (16x32 bf16): p = per-lane row base (already includes hi?+8 dwords),
// dw = k0>>1. Two 16B b128 loads.
__device__ __forceinline__ void ldfragA(bf16x16& f, const uint32_t* p, int dw) {
    u32x4* u = reinterpret_cast<u32x4*>(&f);
    u[0] = *reinterpret_cast<const u32x4*>(p + dw);
    u[1] = *reinterpret_cast<const u32x4*>(p + dw + 4);
}
// B-fragment (32x16 bf16) from Bt[n][k]: p = per-lane column base (includes hi?+8 dwords).
__device__ __forceinline__ void ldfragB(bf16x16& f, const uint32_t* p, int dw) {
    u32x4* u = reinterpret_cast<u32x4*>(&f);
    u[0] = *reinterpret_cast<const u32x4*>(p + dw);
    u[1] = *reinterpret_cast<const u32x4*>(p + dw + 4);
}
// B-fragment read from an LDS-staged tile (64 cols x 32 k, fragment order).
__device__ __forceinline__ void ldfragB_lds(bf16x16& f, const uint32_t* sm, int col, int hi) {
    u32x4* u = reinterpret_cast<u32x4*>(&f);
    const uint32_t* p = sm + col * 16 + (hi ? 8 : 0);
    u[0] = *reinterpret_cast<const u32x4*>(p);
    u[1] = *reinterpret_cast<const u32x4*>(p + 4);
}

// Async-stage one 64col x 32k B tile into LDS: 128 threads x 2 b128 copies.
// LDS byte layout = col*64 + seg*16 (exactly the fragment read pattern above).
__device__ __forceinline__ void fillB_async(const uint16_t* Bt, int n0, int K, int k0,
                                            uint32_t* smbuf, int tid) {
#pragma unroll
    for (int i = 0; i < 2; ++i) {
        const int idx = tid + 128 * i;              // 0..255
        const int col = idx >> 2;
        const int seg = idx & 3;
        const uint64_t ga = (uint64_t)(uintptr_t)(Bt + (size_t)(n0 + col) * K + k0 + seg * 8);
        const uint32_t la = (uint32_t)(uintptr_t)(smbuf + idx * 4);   // LDS byte offset
        asm volatile("global_load_async_to_lds_b128 %0, %1, off"
                     :: "v"(la), "v"(ga) : "memory");
    }
}
__device__ __forceinline__ void wait_async0() {
    asm volatile("s_wait_asynccnt 0x0" ::: "memory");
}

// ---------------- elementwise / layout kernels ----------------
__global__ void cvt_bf16_kernel(const float* __restrict__ in,
                                uint16_t* __restrict__ out, int n) {
    int i = blockIdx.x * blockDim.x + threadIdx.x;
    if (i < n) out[i] = f2bf(in[i]);
}

// Wt[n*K + k] = W[k*N + n]   (W is K x N fp32; Wt is N x K bf16)
__global__ void cvtT_kernel(const float* __restrict__ in,
                            uint16_t* __restrict__ out, int K, int N) {
    int i = blockIdx.x * blockDim.x + threadIdx.x;
    if (i >= K * N) return;
    int k = i / N, n = i % N;
    out[(size_t)n * K + k] = f2bf(in[i]);
}

// Vt[((b*H+h)*DH + d)*NY + key] = Vbf[(b*NY+key)*DIM + h*DH + d]
__global__ void vT_kernel(const uint16_t* __restrict__ Vbf,
                          uint16_t* __restrict__ Vt, int n) {
    int i = blockIdx.x * blockDim.x + threadIdx.x;
    if (i >= n) return;
    int b   = i / (NYc * DIMc);
    int rem = i - b * (NYc * DIMc);
    int key = rem / DIMc;
    int dal = rem - key * DIMc;
    int h = dal / DHc, d = dal - h * DHc;
    Vt[(((size_t)b * Hc + h) * DHc + d) * NYc + key] = Vbf[i];
}

// jax bool mask -> additive fp32 bias (0 / -inf). jax bool_ is 1 byte.
__global__ void maskbias_kernel(const unsigned char* __restrict__ mask,
                                float* __restrict__ mf, int n) {
    int i = blockIdx.x * blockDim.x + threadIdx.x;
    if (i < n) mf[i] = mask[i] ? NEG_INF : 0.0f;
}

// ---------------- WMMA GEMM: C[MxN] = A[MxK]*Bt[NxK]^T + bias ----------------
// block = 128 threads (4 waves); wave w computes a 32x64 tile (2x4 WMMA tiles).
// B tile staged through LDS double buffer with global_load_async_to_lds_b128.
// Requires K % 64 == 0.
__global__ __launch_bounds__(128)
void gemm_bf16_kernel(const uint16_t* __restrict__ A,
                      const uint16_t* __restrict__ Bt,
                      const float* __restrict__ bias,
                      float* __restrict__ Cf,       // fp32 out (or null)
                      uint16_t* __restrict__ Cb,    // bf16 out (or null)
                      int M, int N, int K) {
    __shared__ alignas(16) uint32_t smB[2][1024];   // 2 x 4KB B tiles
    const int tid  = threadIdx.x;
    const int lane = tid & 31;
    const int wave = tid >> 5;
    const int hi   = lane >> 4;
    const int l15  = lane & 15;
    const int m0 = (blockIdx.x * 4 + wave) * 32;
    const int n0 = blockIdx.y * 64;
    const uint32_t* A32 = (const uint32_t*)A;

    // per-lane A row base pointers (K-offset becomes the only loop-variant index)
    const uint32_t* ap0 = A32 + (((size_t)(m0 + l15) * K) >> 1) + (hi ? 8 : 0);
    const uint32_t* ap1 = A32 + (((size_t)(m0 + 16 + l15) * K) >> 1) + (hi ? 8 : 0);

    v8f acc[2][4];
#pragma unroll
    for (int i = 0; i < 2; ++i)
#pragma unroll
        for (int j = 0; j < 4; ++j)
#pragma unroll
            for (int e = 0; e < 8; ++e) acc[i][j][e] = 0.0f;

    bf16x16 aX[2], aY[2], bX[4], bY[4];

    // prologue: stage first B tile, load first A frags
    fillB_async(Bt, n0, K, 0, &smB[0][0], tid);
    ldfragA(aX[0], ap0, 0);
    ldfragA(aX[1], ap1, 0);
    wait_async0();
    __syncthreads();

    for (int k0 = 0; k0 < K; k0 += 64) {
        const bool last = (k0 + 64 >= K);
        // stage B tile for k0+32 while computing k0
        fillB_async(Bt, n0, K, k0 + 32, &smB[1][0], tid);
        ldfragA(aY[0], ap0, (k0 + 32) >> 1);
        ldfragA(aY[1], ap1, (k0 + 32) >> 1);
#pragma unroll
        for (int j = 0; j < 4; ++j) ldfragB_lds(bX[j], &smB[0][0], 16 * j + l15, hi);
#pragma unroll
        for (int i = 0; i < 2; ++i)
#pragma unroll
            for (int j = 0; j < 4; ++j) acc[i][j] = wmma_bf16(aX[i], bX[j], acc[i][j]);
        wait_async0();          // smB[1] writes landed (fragments already consumed smB[0])
        __syncthreads();

        if (!last) {
            fillB_async(Bt, n0, K, k0 + 64, &smB[0][0], tid);
            ldfragA(aX[0], ap0, (k0 + 64) >> 1);
            ldfragA(aX[1], ap1, (k0 + 64) >> 1);
        }
#pragma unroll
        for (int j = 0; j < 4; ++j) ldfragB_lds(bY[j], &smB[1][0], 16 * j + l15, hi);
#pragma unroll
        for (int i = 0; i < 2; ++i)
#pragma unroll
            for (int j = 0; j < 4; ++j) acc[i][j] = wmma_bf16(aY[i], bY[j], acc[i][j]);
        if (!last) {
            wait_async0();
            __syncthreads();
        }
    }

#pragma unroll
    for (int i = 0; i < 2; ++i)
#pragma unroll
        for (int j = 0; j < 4; ++j) {
            const int n = n0 + 16 * j + l15;
            const float bv = bias ? bias[n] : 0.0f;
#pragma unroll
            for (int v = 0; v < 8; ++v) {
                const int m = m0 + 16 * i + v + 8 * hi;   // C layout: VGPR v -> M=v (+8 hi)
                const float val = acc[i][j][v] + bv;
                const size_t idx = (size_t)m * N + n;
                if (Cf) Cf[idx] = val;
                if (Cb) Cb[idx] = f2bf(val);
            }
        }
}

// ---------------- flash attention (transposed): S^T = Kh*Qh^T, O^T = Vh^T*P^T ----------
// grid = (NX/64, H, B), block = 128 (4 waves, one 16-query tile each)
__global__ __launch_bounds__(128)
void attn_kernel(const uint16_t* __restrict__ Qbf,   // [B,NX,DIM] bf16
                 const uint16_t* __restrict__ Kbf,   // [B,NY,DIM] bf16
                 const uint16_t* __restrict__ Vt,    // [B,H,DH,NY] bf16
                 const float*    __restrict__ maskf, // [B,NY] additive bias
                 float* __restrict__ Oattn) {        // [B,NX,DIM] fp32
    const int b = blockIdx.z, h = blockIdx.y;
    const int lane = threadIdx.x & 31;
    const int wave = threadIdx.x >> 5;
    const int hi  = lane >> 4;
    const int l15 = lane & 15;
    const int qbase = blockIdx.x * 64 + wave * 16;
    const float scale = 0.03125f;                    // 1/sqrt(1024)

    const uint32_t* Q32 = (const uint32_t*)Qbf + (((size_t)b * NXc * DIMc + h * DHc) >> 1);
    const uint32_t* K32 = (const uint32_t*)Kbf + (((size_t)b * NYc * DIMc + h * DHc) >> 1);
    const uint32_t* V32 = (const uint32_t*)Vt  + (((size_t)(b * Hc + h) * DHc * NYc) >> 1);
    const float* mrow = maskf + (size_t)b * NYc;

    // per-lane base pointers; all intra-tile offsets are compile-time immediates
    const uint32_t* qp = Q32 + (((size_t)(qbase + l15) * DIMc) >> 1) + (hi ? 8 : 0); // B-frag
    const uint32_t* kp = K32 + (((size_t)l15 * DIMc) >> 1) + (hi ? 8 : 0);           // A-frag
    const uint32_t* vp = V32 + (((size_t)l15 * NYc) >> 1) + (hi ? 8 : 0);            // A-frag

    // Qh^T B-fragments (K = head-dim), resident for whole key loop
    bf16x16 qf[4];
#pragma unroll
    for (int d = 0; d < 4; ++d) ldfragB(qf[d], qp, d * 16);

    v8f o[8];                                        // O^T accum: 128(d) x 16(q)
#pragma unroll
    for (int t = 0; t < 8; ++t)
#pragma unroll
        for (int e = 0; e < 8; ++e) o[t][e] = 0.0f;

    float m_run = NEG_INF, l_run = 0.0f;

    for (int kb = 0; kb < NYc; kb += 32) {
        const uint32_t* kbp = kp + (size_t)kb * (DIMc / 2);   // rows kb+l15
        const uint32_t* vbp = vp + (kb >> 1);                 // key offset in Vt rows
        // issue ALL loads up front: K-frags for S, then V-frags (independent of S)
        bf16x16 ka0[4], ka1[4];
#pragma unroll
        for (int d = 0; d < 4; ++d) {
            ldfragA(ka0[d], kbp, d * 16);
            ldfragA(ka1[d], kbp + 16 * (DIMc / 2), d * 16);   // rows kb+16+l15
        }
        bf16x16 va[8];
#pragma unroll
        for (int t = 0; t < 8; ++t)
            ldfragA(va[t], vbp + (size_t)t * 16 * (NYc / 2), 0);

        // S^T tiles: s0 = keys kb..kb+15, s1 = keys kb+16..kb+31 (C layout: N = query)
        v8f s0, s1;
#pragma unroll
        for (int e = 0; e < 8; ++e) { s0[e] = 0.0f; s1[e] = 0.0f; }
#pragma unroll
        for (int d = 0; d < 4; ++d) {
            s0 = wmma_bf16(ka0[d], qf[d], s0);
            s1 = wmma_bf16(ka1[d], qf[d], s1);
        }

        // scale + mask bias; per-lane row (=query) statistics
        float e0[8], e1[8];
        float tmax = NEG_INF;
#pragma unroll
        for (int v = 0; v < 8; ++v) {
            const float x0 = s0[v] * scale + mrow[kb + 8 * hi + v];        // key = v+8*hi
            const float x1 = s1[v] * scale + mrow[kb + 16 + 8 * hi + v];
            e0[v] = x0; e1[v] = x1;
            tmax = fmaxf(tmax, fmaxf(x0, x1));
        }
        tmax = fmaxf(tmax, __shfl_xor(tmax, 16, 32));       // merge lane halves
        const float mnew = fmaxf(m_run, tmax);
        const bool dead = (mnew == NEG_INF);                // fully-masked so far
        const float factor = dead ? 1.0f : __expf(m_run - mnew);

        float lsum = 0.0f;
#pragma unroll
        for (int v = 0; v < 8; ++v) {
            e0[v] = dead ? 0.0f : __expf(e0[v] - mnew);
            e1[v] = dead ? 0.0f : __expf(e1[v] - mnew);
            lsum += e0[v] + e1[v];
        }
        lsum += __shfl_xor(lsum, 16, 32);
        l_run = l_run * factor + lsum;
        m_run = mnew;

        // build P^T B-fragment (32 keys x 16 queries): swap lane halves, pack bf16
        float o0[8], o1[8];
#pragma unroll
        for (int v = 0; v < 8; ++v) {
            o0[v] = __shfl_xor(e0[v], 16, 32);
            o1[v] = __shfl_xor(e1[v], 16, 32);
        }
        bf16x16 pfrag;
        uint32_t* pu = reinterpret_cast<uint32_t*>(&pfrag);
#pragma unroll
        for (int v = 0; v < 4; ++v) {
            // lo lanes: keys 2v,2v+1 (own e0) ; hi lanes: keys 16+2v (partner e1)
            const float a0 = hi ? o1[2 * v]     : e0[2 * v];
            const float a1 = hi ? o1[2 * v + 1] : e0[2 * v + 1];
            pu[v] = pack_bf2(a0, a1);
            // lo lanes: keys 8+2v (partner e0) ; hi lanes: keys 24+2v (own e1)
            const float b0 = hi ? e1[2 * v]     : o0[2 * v];
            const float b1 = hi ? e1[2 * v + 1] : o0[2 * v + 1];
            pu[4 + v] = pack_bf2(b0, b1);
        }

        // O^T += V^T(d x keys) * P^T(keys x q); rescale accum by `factor` first
#pragma unroll
        for (int t = 0; t < 8; ++t) {
#pragma unroll
            for (int e = 0; e < 8; ++e) o[t][e] *= factor;
            o[t] = wmma_bf16(va[t], pfrag, o[t]);
        }
    }

    // normalize (fully-masked query -> 0) and scatter O^T back to [q][d]
    const float rcp = (l_run > 0.0f) ? (1.0f / l_run) : 0.0f;
    const size_t orow = ((size_t)b * NXc + qbase + l15) * DIMc + h * DHc;
#pragma unroll
    for (int t = 0; t < 8; ++t)
#pragma unroll
        for (int v = 0; v < 8; ++v) {
            const int d = t * 16 + v + 8 * hi;
            Oattn[orow + d] = o[t][v] * rcp;
        }
}

// ---------------- fused residual (+ReLU) + LayerNorm ----------------
// out = LN(A + (relu_b ? relu(Bv) : Bv)) * g + b  ; one row per block
__global__ __launch_bounds__(256)
void addln_kernel(const float* __restrict__ A, const float* __restrict__ Bv,
                  const float* __restrict__ g, const float* __restrict__ bb,
                  float* __restrict__ outf, uint16_t* __restrict__ outb,
                  int relu_b) {
    const int row = blockIdx.x;
    const int tid = threadIdx.x;
    const size_t base = (size_t)row * DIMc;
    __shared__ float red[256];

    float x[4];
    float s = 0.0f;
#pragma unroll
    for (int i = 0; i < 4; ++i) {
        const int idx = tid + 256 * i;
        float v = Bv[base + idx];
        if (relu_b) v = fmaxf(v, 0.0f);
        v += A[base + idx];
        x[i] = v; s += v;
    }
    red[tid] = s; __syncthreads();
    for (int st = 128; st > 0; st >>= 1) {
        if (tid < st) red[tid] += red[tid + st];
        __syncthreads();
    }
    const float mu = red[0] * (1.0f / DIMc);
    __syncthreads();

    float sv = 0.0f;
#pragma unroll
    for (int i = 0; i < 4; ++i) { const float d = x[i] - mu; sv += d * d; }
    red[tid] = sv; __syncthreads();
    for (int st = 128; st > 0; st >>= 1) {
        if (tid < st) red[tid] += red[tid + st];
        __syncthreads();
    }
    const float rstd = rsqrtf(red[0] * (1.0f / DIMc) + 1e-5f);

#pragma unroll
    for (int i = 0; i < 4; ++i) {
        const int idx = tid + 256 * i;
        const float y = (x[i] - mu) * rstd * g[idx] + bb[idx];
        outf[base + idx] = y;
        if (outb) outb[base + idx] = f2bf(y);
    }
}

// ---------------- launch ----------------
extern "C" void kernel_launch(void* const* d_in, const int* in_sizes, int n_in,
                              void* d_out, int out_size, void* d_ws, size_t ws_size,
                              hipStream_t stream) {
    const float* X    = (const float*)d_in[0];
    const float* Y    = (const float*)d_in[1];
    const unsigned char* mask = (const unsigned char*)d_in[2];  // jax bool = 1 byte
    const float* Wq = (const float*)d_in[3];  const float* bq = (const float*)d_in[4];
    const float* Wk = (const float*)d_in[5];  const float* bk = (const float*)d_in[6];
    const float* Wv = (const float*)d_in[7];  const float* bv = (const float*)d_in[8];
    const float* Wo = (const float*)d_in[9];  const float* bo = (const float*)d_in[10];
    const float* g1 = (const float*)d_in[11]; const float* b1 = (const float*)d_in[12];
    const float* g2 = (const float*)d_in[13]; const float* b2 = (const float*)d_in[14];
    float* out = (float*)d_out;

    uint8_t* ws = (uint8_t*)d_ws;
    size_t off = 0;
    auto alloc = [&](size_t bytes) -> void* {
        void* p = ws + off;
        off = (off + bytes + 255) & ~(size_t)255;
        return p;
    };
    const size_t NE  = (size_t)Bc * NXc * DIMc;      // 8M elements
    uint16_t* Xbf = (uint16_t*)alloc(NE * 2);
    uint16_t* Ybf = (uint16_t*)alloc(NE * 2);
    uint16_t* WqT = (uint16_t*)alloc((size_t)DIMc * DIMc * 2);
    uint16_t* WkT = (uint16_t*)alloc((size_t)DIMc * DIMc * 2);
    uint16_t* WvT = (uint16_t*)alloc((size_t)DIMc * DIMc * 2);
    uint16_t* WoT = (uint16_t*)alloc((size_t)DIMc * DIMc * 2);
    float*    Qf  = (float*)   alloc(NE * 4);
    uint16_t* Qbf = (uint16_t*)alloc(NE * 2);
    uint16_t* Kbf = (uint16_t*)alloc(NE * 2);
    uint16_t* Vbf = (uint16_t*)alloc(NE * 2);
    uint16_t* Vt  = (uint16_t*)alloc(NE * 2);
    float*    mbf = (float*)   alloc((size_t)Bc * NYc * 4);
    float*    Att = (float*)   alloc(NE * 4);
    float*    Zf  = (float*)   alloc(NE * 4);
    uint16_t* Zbf = (uint16_t*)alloc(NE * 2);
    float*    Rf  = (float*)   alloc(NE * 4);

    const int n8M = (int)NE;
    const int TPB = 256;
    const int grid8M = (n8M + TPB - 1) / TPB;
    const int grid1M = (DIMc * DIMc + TPB - 1) / TPB;

    // stage precision: fp32 -> bf16 (WMMA path), weights transposed for B-frag loads
    cvt_bf16_kernel<<<grid8M, TPB, 0, stream>>>(X, Xbf, n8M);
    cvt_bf16_kernel<<<grid8M, TPB, 0, stream>>>(Y, Ybf, n8M);
    cvtT_kernel<<<grid1M, TPB, 0, stream>>>(Wq, WqT, DIMc, DIMc);
    cvtT_kernel<<<grid1M, TPB, 0, stream>>>(Wk, WkT, DIMc, DIMc);
    cvtT_kernel<<<grid1M, TPB, 0, stream>>>(Wv, WvT, DIMc, DIMc);
    cvtT_kernel<<<grid1M, TPB, 0, stream>>>(Wo, WoT, DIMc, DIMc);
    maskbias_kernel<<<(Bc * NYc + TPB - 1) / TPB, TPB, 0, stream>>>(mask, mbf, Bc * NYc);

    // QKV projections (bf16 WMMA, fp32 accum); 32x64 tile per wave, LDS-staged B
    dim3 ggrid(Mrows / 128, DIMc / 64);
    gemm_bf16_kernel<<<ggrid, 128, 0, stream>>>(Xbf, WqT, bq, Qf, Qbf, Mrows, DIMc, DIMc);
    gemm_bf16_kernel<<<ggrid, 128, 0, stream>>>(Ybf, WkT, bk, nullptr, Kbf, Mrows, DIMc, DIMc);
    gemm_bf16_kernel<<<ggrid, 128, 0, stream>>>(Ybf, WvT, bv, nullptr, Vbf, Mrows, DIMc, DIMc);
    vT_kernel<<<grid8M, TPB, 0, stream>>>(Vbf, Vt, n8M);

    // flash attention
    attn_kernel<<<dim3(NXc / 64, Hc, Bc), 128, 0, stream>>>(Qbf, Kbf, Vt, mbf, Att);

    // LN1: Z = LN(Q + attn); also emit bf16 for the Wo GEMM
    addln_kernel<<<Mrows, 256, 0, stream>>>(Qf, Att, g1, b1, Zf, Zbf, 0);

    // output projection R = Z @ Wo + bo
    gemm_bf16_kernel<<<ggrid, 128, 0, stream>>>(Zbf, WoT, bo, Rf, nullptr, Mrows, DIMc, DIMc);

    // LN2: out = LN(Z + relu(R))
    addln_kernel<<<Mrows, 256, 0, stream>>>(Zf, Rf, g2, b2, out, nullptr, 1);

    (void)in_sizes; (void)n_in; (void)out_size; (void)ws_size;
}
